// SelfAttention_2276332667249
// MI455X (gfx1250) — matrix operand order repaired and checked by reading, not compiled, not run on hardware
//
#include <hip/hip_runtime.h>
#include <hip/hip_bf16.h>

// ---------------------------------------------------------------------------
// Self-attention (B=2, L=2048, C=1024, H=16, D=64) for gfx1250 (MI455X).
// All GEMM-shaped work goes through v_wmma_f32_16x16x32_bf16 (wave32 WMMA).
// GEMM A-tiles staged with global_load_async_to_lds_b128 (ASYNCcnt path).
// ---------------------------------------------------------------------------

constexpr int cB = 2;
constexpr int cL = 2048;
constexpr int cC = 1024;
constexpr int cH = 16;
constexpr int cD = 64;
constexpr int cM = cB * cL;      // 4096 token rows

typedef __attribute__((ext_vector_type(16))) __bf16 v16bf;
typedef __attribute__((ext_vector_type(8)))  float  v8f;

union FragB {
    v16bf v;
    uint4 q[2];
    unsigned short u[16];
};

__device__ __forceinline__ unsigned short f2bf(float f) {
    unsigned int x = __float_as_uint(f);
    unsigned int r = x + 0x7FFFu + ((x >> 16) & 1u);   // round-to-nearest-even
    return (unsigned short)(r >> 16);
}

__device__ __forceinline__ v8f wmma_bf16(const FragB& a, const FragB& b, v8f c) {
    return __builtin_amdgcn_wmma_f32_16x16x32_bf16(
        /*neg_a=*/false, a.v, /*neg_b=*/false, b.v,
        /*c_mod=*/(short)0, c, /*reuse_a=*/false, /*reuse_b=*/false);
}

// Issue an async memory->LDS 128-bit copy (per-lane addresses).
// LDS offset = low 32 bits of the flat shared-pointer address (flat->LDS
// aperture truncation, ISA 10.2); tracked by ASYNCcnt.
__device__ __forceinline__ void async_g2l_b128(const void* gptr, void* lptr) {
    unsigned lds = (unsigned)(unsigned long long)lptr;
    asm volatile("global_load_async_to_lds_b128 %0, %1, off"
                 :: "v"(lds), "v"((unsigned long long)gptr)
                 : "memory");
}
__device__ __forceinline__ void wait_async0() {
    asm volatile("s_wait_asynccnt 0x0" ::: "memory");
}

// ---------------------------------------------------------------------------
// fp32 -> bf16 bulk convert (n must be a multiple of 4)
// ---------------------------------------------------------------------------
__global__ __launch_bounds__(256) void cvt_f32_bf16(
    const float* __restrict__ in, unsigned short* __restrict__ out, int n4) {
    int i = blockIdx.x * blockDim.x + threadIdx.x;
    if (i < n4) {
        float4 v = ((const float4*)in)[i];
        uint2 pk;
        pk.x = (unsigned)f2bf(v.x) | ((unsigned)f2bf(v.y) << 16);
        pk.y = (unsigned)f2bf(v.z) | ((unsigned)f2bf(v.w) << 16);
        ((uint2*)out)[i] = pk;
    }
}

// ---------------------------------------------------------------------------
// bf16 GEMM: C[M,N] (f32) = A[M,K] (bf16) * B[K,N] (bf16); M,N,K compile-time
// so every address is shifts + immediate offsets. Block tile 128x128, K-step
// 32. 8 waves in 2x4; each wave owns a 64x32 tile = 4x2 WMMA accumulators.
// A-tile: async global->LDS b128 copies (ASYNCcnt). B-tile: transposed [n][k]
// scatter so every B fragment is contiguous b128 LDS reads.
// ---------------------------------------------------------------------------
#define TM 128
#define TN 128
#define TK 32

template <int M, int N, int K>
__global__ __launch_bounds__(256) void gemm_bf16_f32(
    const unsigned short* __restrict__ A,
    const unsigned short* __restrict__ B,
    float* __restrict__ C)
{
    __shared__ __align__(16) unsigned short As[TM * TK];
    __shared__ __align__(16) unsigned short Bs[TN * TK];

    const int t    = threadIdx.x;
    const int lane = t & 31;
    const int wid  = t >> 5;
    const int ln   = lane & 15;
    const int lg   = lane >> 4;
    const int m0   = blockIdx.y * TM;
    const int n0   = blockIdx.x * TN;
    const int wm   = (wid >> 2) * 64;   // wave M offset: 0 / 64
    const int wn   = (wid & 3) * 32;    // wave N offset: 0/32/64/96

    v8f acc[4][2];
#pragma unroll
    for (int i = 0; i < 4; ++i)
#pragma unroll
        for (int j = 0; j < 2; ++j)
#pragma unroll
            for (int e = 0; e < 8; ++e) acc[i][j][e] = 0.0f;

    for (int kk = 0; kk < K; kk += TK) {
        // --- A tile: 128 rows x 32 k = 512 async b128 copies, no VGPR stage
#pragma unroll
        for (int it = 0; it < 2; ++it) {
            int idx = t + it * 256;
            int row = idx >> 2;
            int kp  = (idx & 3) * 8;
            async_g2l_b128(A + (size_t)(m0 + row) * K + kk + kp,
                           &As[row * TK + kp]);
        }
        // --- B tile transposed: [n][k] ---
#pragma unroll
        for (int it = 0; it < 2; ++it) {
            int idx = t + it * 256;
            int kr  = idx >> 4;          // 0..31
            int nc  = (idx & 15) * 8;    // 0..120
            uint4 bv = *(const uint4*)(B + (size_t)(kk + kr) * N + n0 + nc);
            const unsigned short* bu = (const unsigned short*)&bv;
#pragma unroll
            for (int e = 0; e < 8; ++e) Bs[(nc + e) * TK + kr] = bu[e];
        }
        // prefetch next K-step tiles into cache while this one computes
        if (kk + TK < K) {
            __builtin_prefetch(A + (size_t)(m0 + (t >> 2)) * K + kk + TK, 0, 1);
            __builtin_prefetch(B + (size_t)(kk + TK + (t >> 4)) * N + n0 +
                                   (t & 15) * 8, 0, 1);
        }
        wait_async0();           // drain ASYNCcnt before waves cross-consume
        __syncthreads();

        FragB a[4], b[2];
#pragma unroll
        for (int i = 0; i < 4; ++i) {
            int row = wm + i * 16 + ln;
            a[i].q[0] = *(const uint4*)&As[row * TK + 8 * lg];
            a[i].q[1] = *(const uint4*)&As[row * TK + 16 + 8 * lg];
        }
#pragma unroll
        for (int j = 0; j < 2; ++j) {
            int col = wn + j * 16 + ln;
            b[j].q[0] = *(const uint4*)&Bs[col * TK + 8 * lg];
            b[j].q[1] = *(const uint4*)&Bs[col * TK + 16 + 8 * lg];
        }
#pragma unroll
        for (int i = 0; i < 4; ++i)
#pragma unroll
            for (int j = 0; j < 2; ++j)
                acc[i][j] = wmma_bf16(a[i], b[j], acc[i][j]);
        __syncthreads();
    }

    // D layout: lane ln+16*lg, elem r -> (m = r + 8*lg, n = ln) within tile.
    // Row stride is compile-time => folds into store immediate offsets.
    float* crow = C + (size_t)(m0 + wm + 8 * lg) * N + n0 + wn + ln;
#pragma unroll
    for (int i = 0; i < 4; ++i)
#pragma unroll
        for (int j = 0; j < 2; ++j)
#pragma unroll
            for (int r = 0; r < 8; ++r)
                crow[(size_t)(i * 16 + r) * N + j * 16] = acc[i][j][r];
}

// ---------------------------------------------------------------------------
// QK-LayerNorm + d^-0.25 scaling + bf16 pack.
// Q,K -> [B,H,L,D] bf16 ; V -> transposed [B,H,D,L] bf16 (so flash-attn V
// B-fragments are contiguous along the key axis).
// One wave per (b,l,h) row; lane handles 2 of the 64 head elements.
// ---------------------------------------------------------------------------
__global__ __launch_bounds__(128) void qkv_pack(
    const float* __restrict__ qkv,          // [B*L, 3C]
    const float* __restrict__ q_scale, const float* __restrict__ q_bias,
    const float* __restrict__ k_scale, const float* __restrict__ k_bias,
    unsigned short* __restrict__ Qo, unsigned short* __restrict__ Ko,
    unsigned short* __restrict__ Vt)
{
    const int lane = threadIdx.x & 31;
    const int wid  = threadIdx.x >> 5;
    const int w    = blockIdx.x * 4 + wid;       // 0 .. B*L*H-1
    const int bl   = w / cH;                     // b*L + l
    const int h    = w % cH;
    const int b    = bl / cL;
    const int l    = bl % cL;
    const int d0   = lane * 2;
    const float inv4 = 0.35355339059327373f;     // 64^-0.25

    const size_t base  = (size_t)bl * (3 * cC) + h * cD;
    const size_t qkrow = ((size_t)(b * cH + h) * cL + l) * cD + d0;

    // ---- Q ----
    {
        float2 v = *(const float2*)(qkv + base + d0);
        float s = v.x + v.y;
#pragma unroll
        for (int off = 16; off >= 1; off >>= 1) s += __shfl_xor(s, off, 32);
        float mean = s * (1.0f / 64.0f);
        float dx = v.x - mean, dy = v.y - mean;
        float vs = dx * dx + dy * dy;
#pragma unroll
        for (int off = 16; off >= 1; off >>= 1) vs += __shfl_xor(vs, off, 32);
        float rstd = rsqrtf(vs * (1.0f / 64.0f) + 1e-6f);
        float ox = (dx * rstd * q_scale[d0]     + q_bias[d0])     * inv4;
        float oy = (dy * rstd * q_scale[d0 + 1] + q_bias[d0 + 1]) * inv4;
        *(unsigned int*)(Qo + qkrow) =
            (unsigned)f2bf(ox) | ((unsigned)f2bf(oy) << 16);
    }
    // ---- K ----
    {
        float2 v = *(const float2*)(qkv + base + cC + d0);
        float s = v.x + v.y;
#pragma unroll
        for (int off = 16; off >= 1; off >>= 1) s += __shfl_xor(s, off, 32);
        float mean = s * (1.0f / 64.0f);
        float dx = v.x - mean, dy = v.y - mean;
        float vs = dx * dx + dy * dy;
#pragma unroll
        for (int off = 16; off >= 1; off >>= 1) vs += __shfl_xor(vs, off, 32);
        float rstd = rsqrtf(vs * (1.0f / 64.0f) + 1e-6f);
        float ox = (dx * rstd * k_scale[d0]     + k_bias[d0])     * inv4;
        float oy = (dy * rstd * k_scale[d0 + 1] + k_bias[d0 + 1]) * inv4;
        *(unsigned int*)(Ko + qkrow) =
            (unsigned)f2bf(ox) | ((unsigned)f2bf(oy) << 16);
    }
    // ---- V (transposed store) ----
    {
        float2 v = *(const float2*)(qkv + base + 2 * cC + d0);
        size_t vbase = ((size_t)(b * cH + h) * cD + d0) * cL + l;
        Vt[vbase]      = f2bf(v.x);
        Vt[vbase + cL] = f2bf(v.y);
    }
}

// ---------------------------------------------------------------------------
// Flash attention per (b,h). Block = 128 threads = 4 independent waves,
// each wave owns 16 query rows; streams 64-key tiles with online softmax.
//   S = Q K^T     : 4 key sub-tiles x (2 chained 16x16x32 WMMAs over D=64)
//   O += P V      : 4 d sub-tiles  x (2 chained WMMAs over 64 keys)
// P is transposed D-layout -> A-layout through wave-private LDS (in-order
// DS pipe; no barrier needed).
// Output written bf16 as [B*L, H*D] — directly the A matrix of the final GEMM.
// ---------------------------------------------------------------------------
__global__ __launch_bounds__(128) void flash_attn(
    const unsigned short* __restrict__ Q,   // [B*H, L, 64]
    const unsigned short* __restrict__ Km,  // [B*H, L, 64]
    const unsigned short* __restrict__ Vt,  // [B*H, 64, L]
    unsigned short* __restrict__ O)         // [B*L, 1024] bf16
{
    __shared__ __align__(16) unsigned short Pl[4][16 * 64];

    const int lane = threadIdx.x & 31;
    const int wid  = threadIdx.x >> 5;
    const int ln   = lane & 15;
    const int lg   = lane >> 4;
    const int nqt  = cL / 64;                    // 32 q-tiles per (b,h)
    const int blk  = blockIdx.x;
    const int bh   = blk / nqt;
    const int qt   = blk % nqt;
    const int b    = bh / cH;
    const int h    = bh % cH;

    const unsigned short* Qb = Q  + (size_t)bh * cL * cD;
    const unsigned short* Kb = Km + (size_t)bh * cL * cD;
    const unsigned short* Vb = Vt + (size_t)bh * cD * cL;

    const int qrow0 = qt * 64 + wid * 16;

    // Preload Q A-fragments (2 chunks of D=32), reused over all key tiles.
    FragB aq[2];
#pragma unroll
    for (int c = 0; c < 2; ++c) {
        const unsigned short* p = Qb + (size_t)(qrow0 + ln) * cD + 32 * c;
        aq[c].q[0] = *(const uint4*)(p + 8 * lg);
        aq[c].q[1] = *(const uint4*)(p + 16 + 8 * lg);
    }

    v8f accO[4];
#pragma unroll
    for (int tt = 0; tt < 4; ++tt)
#pragma unroll
        for (int e = 0; e < 8; ++e) accO[tt][e] = 0.0f;

    float mi[8], li[8];
#pragma unroll
    for (int r = 0; r < 8; ++r) { mi[r] = -1e30f; li[r] = 0.0f; }

    for (int kt = 0; kt < cL; kt += 64) {
        // prefetch next key/value tiles (global_prefetch_b8)
        if (kt + 64 < cL) {
            __builtin_prefetch(Kb + (size_t)(kt + 64 + lane) * cD, 0, 1);
            __builtin_prefetch(Kb + (size_t)(kt + 96 + lane) * cD, 0, 1);
            __builtin_prefetch(Vb + (size_t)(lane * 2) * cL + kt + 64, 0, 1);
            __builtin_prefetch(Vb + (size_t)(lane * 2 + 1) * cL + kt + 64, 0, 1);
        }

        // ---- S = Q K^T for this 64-key tile ----
        v8f S[4];
#pragma unroll
        for (int j = 0; j < 4; ++j) {
            v8f s;
#pragma unroll
            for (int e = 0; e < 8; ++e) s[e] = 0.0f;
#pragma unroll
            for (int c = 0; c < 2; ++c) {
                FragB bk;
                const unsigned short* p =
                    Kb + (size_t)(kt + j * 16 + ln) * cD + 32 * c + 8 * lg;
                bk.q[0] = *(const uint4*)(p);
                bk.q[1] = *(const uint4*)(p + 16);
                s = wmma_bf16(aq[c], bk, s);
            }
            S[j] = s;
        }

        // ---- online softmax (row = r + 8*lg; reduce over 16-lane group) ----
        float mnew[8], alpha[8];
#pragma unroll
        for (int r = 0; r < 8; ++r) {
            float mx = fmaxf(fmaxf(S[0][r], S[1][r]), fmaxf(S[2][r], S[3][r]));
            mx = fmaxf(mx, __shfl_xor(mx, 1, 16));
            mx = fmaxf(mx, __shfl_xor(mx, 2, 16));
            mx = fmaxf(mx, __shfl_xor(mx, 4, 16));
            mx = fmaxf(mx, __shfl_xor(mx, 8, 16));
            mnew[r]  = fmaxf(mi[r], mx);
            alpha[r] = __expf(mi[r] - mnew[r]);
            mi[r]    = mnew[r];
        }
        float rs[8];
#pragma unroll
        for (int r = 0; r < 8; ++r) rs[r] = 0.0f;
#pragma unroll
        for (int j = 0; j < 4; ++j)
#pragma unroll
            for (int r = 0; r < 8; ++r) {
                float p = __expf(S[j][r] - mnew[r]);
                S[j][r] = p;
                rs[r] += p;
            }
#pragma unroll
        for (int r = 0; r < 8; ++r) {
            float s = rs[r];
            s += __shfl_xor(s, 1, 16);
            s += __shfl_xor(s, 2, 16);
            s += __shfl_xor(s, 4, 16);
            s += __shfl_xor(s, 8, 16);
            li[r] = li[r] * alpha[r] + s;
        }
#pragma unroll
        for (int tt = 0; tt < 4; ++tt)
#pragma unroll
            for (int r = 0; r < 8; ++r) accO[tt][r] *= alpha[r];

        // ---- transpose P (D layout -> A layout) via wave-private LDS ----
#pragma unroll
        for (int j = 0; j < 4; ++j)
#pragma unroll
            for (int r = 0; r < 8; ++r)
                Pl[wid][(r + 8 * lg) * 64 + j * 16 + ln] = f2bf(S[j][r]);

        FragB ap[2];
#pragma unroll
        for (int c = 0; c < 2; ++c) {
            const unsigned short* p = &Pl[wid][ln * 64 + 32 * c];
            ap[c].q[0] = *(const uint4*)(p + 8 * lg);
            ap[c].q[1] = *(const uint4*)(p + 16 + 8 * lg);
        }

        // ---- O += P V  (V fragments contiguous thanks to [d][l] layout) ----
#pragma unroll
        for (int tt = 0; tt < 4; ++tt)
#pragma unroll
            for (int c = 0; c < 2; ++c) {
                FragB bv;
                const unsigned short* p =
                    Vb + (size_t)(tt * 16 + ln) * cL + kt + 32 * c + 8 * lg;
                bv.q[0] = *(const uint4*)(p);
                bv.q[1] = *(const uint4*)(p + 16);
                accO[tt] = wmma_bf16(ap[c], bv, accO[tt]);
            }
    }

    // ---- epilogue: O / l, bf16, GEMM-ready layout [b*L+q, h*64+d] ----
#pragma unroll
    for (int tt = 0; tt < 4; ++tt)
#pragma unroll
        for (int r = 0; r < 8; ++r) {
            float v = accO[tt][r] / li[r];
            int qrow = qrow0 + r + 8 * lg;
            size_t row = (size_t)b * cL + qrow;
            O[row * cC + h * cD + tt * 16 + ln] = f2bf(v);
        }
}

// ---------------------------------------------------------------------------
// Host-side orchestration (graph-capture safe: stream-only, no allocs).
// ---------------------------------------------------------------------------
extern "C" void kernel_launch(void* const* d_in, const int* in_sizes, int n_in,
                              void* d_out, int out_size, void* d_ws, size_t ws_size,
                              hipStream_t stream) {
    const float* x       = (const float*)d_in[0];
    const float* w_qkv   = (const float*)d_in[1];
    const float* w_out   = (const float*)d_in[2];
    const float* q_scale = (const float*)d_in[3];
    const float* q_bias  = (const float*)d_in[4];
    const float* k_scale = (const float*)d_in[5];
    const float* k_bias  = (const float*)d_in[6];
    float* out = (float*)d_out;

    // workspace carve (~101 MB)
    char* ws = (char*)d_ws;
    auto carve = [&](size_t bytes) {
        char* p = ws;
        ws += (bytes + 255) & ~(size_t)255;
        return p;
    };
    unsigned short* xb    = (unsigned short*)carve((size_t)cM * cC * 2);          // x bf16
    unsigned short* wqkvb = (unsigned short*)carve((size_t)cC * 3 * cC * 2);      // w_qkv bf16
    unsigned short* woutb = (unsigned short*)carve((size_t)cC * cC * 2);          // w_out bf16
    float*          qkvf  = (float*)carve((size_t)cM * 3 * cC * 4);               // qkv fp32
    unsigned short* qb    = (unsigned short*)carve((size_t)cB * cH * cL * cD * 2);
    unsigned short* kb    = (unsigned short*)carve((size_t)cB * cH * cL * cD * 2);
    unsigned short* vt    = (unsigned short*)carve((size_t)cB * cH * cD * cL * 2);
    unsigned short* attnb = (unsigned short*)carve((size_t)cM * cC * 2);          // attn out bf16

    // 1) converts
    {
        int n4 = cM * cC / 4;
        cvt_f32_bf16<<<(n4 + 255) / 256, 256, 0, stream>>>(x, xb, n4);
    }
    {
        int n4 = cC * 3 * cC / 4;
        cvt_f32_bf16<<<(n4 + 255) / 256, 256, 0, stream>>>(w_qkv, wqkvb, n4);
    }
    {
        int n4 = cC * cC / 4;
        cvt_f32_bf16<<<(n4 + 255) / 256, 256, 0, stream>>>(w_out, woutb, n4);
    }

    // 2) QKV projection: [4096,1024] x [1024,3072] -> fp32
    {
        dim3 grid(3 * cC / TN, cM / TM);
        gemm_bf16_f32<cM, 3 * cC, cC><<<grid, 256, 0, stream>>>(xb, wqkvb, qkvf);
    }

    // 3) QK layernorm + scale + pack, V transpose
    {
        int waves = cB * cL * cH;            // one wave per (b,l,h)
        qkv_pack<<<waves / 4, 128, 0, stream>>>(qkvf, q_scale, q_bias,
                                                k_scale, k_bias, qb, kb, vt);
    }

    // 4) flash attention
    {
        int blocks = cB * cH * (cL / 64);    // 1024
        flash_attn<<<blocks, 128, 0, stream>>>(qb, kb, vt, attnb);
    }

    // 5) output projection: [4096,1024] x [1024,1024] -> fp32 d_out
    {
        dim3 grid(cC / TN, cM / TM);
        gemm_bf16_f32<cM, cC, cC><<<grid, 256, 0, stream>>>(attnb, woutb, out);
    }
}
